// TripleCompressor_78099685310979
// MI455X (gfx1250) — compile-verified, matrix operand order checked by reading, not run on hardware
//
// TripleCompressor for MI455X (gfx1250, wave32, WMMA).
//
// Roofline (round 2): the bottleneck is L2 *weight* bandwidth, not FLOPs:
// per-block weight stream is ~3.4MB bf16; with 1 seq/block that is ~83GB of
// L2 reads vs ~0.3ms of WMMA math and ~13us of HBM. Fix: batch G=4 sequences
// per workgroup and reuse each WMMA B-fragment across 4 A-tiles (4 accum
// sets), cutting L2 weight traffic 4x. QKV/FFN-hidden scratch is bf16 in LDS
// (it feeds the next WMMA as bf16 anyway; ReLU+bias fused into writeback);
// the residual stays f32. LDS ~236KB -> 1 block (8 waves) per WGP.
#include <hip/hip_runtime.h>
#include <stdint.h>

#define BDIM   1024
#define MDIM   8
#define RDIM   3
#define SDIM   12
#define DDIM   256
#define HHEADS 4
#define DHEAD  64
#define FFDIM  1024
#define NSEQ   (BDIM * MDIM * RDIM)
#define GSEQ   4
#define NTHR   256
#define NWAVES 8
#define EPSV   1e-5f
#define NEGV   -1e9f

typedef __bf16 bf16;
typedef __attribute__((ext_vector_type(16))) __bf16 v16bf;
typedef __attribute__((ext_vector_type(8)))  float  v8f;

// -------- workspace layout (bf16 elements) --------
#define WS_WQKV   0                      // 768*256
#define WS_WO     196608                 // 256*256
#define WS_W1     262144                 // 1024*256
#define WS_W2     524288                 // 256*1024
#define WS_LAYER  786432                 // per-layer stride
#define WS_CWKV   (2 * WS_LAYER)         // 512*256 (cross K/V rows of cross_Wqkv)

struct SeqParams {
  const int*     token_ids;
  const uint8_t* tok_mask;     // (NSEQ, S)  nonzero = pad
  const uint8_t* trip_mask;    // (B, M)     nonzero = pad
  const float*   token_emb;
  const float*   token_pos;
  const float *ln1_g[2], *ln1_b[2], *bqkv[2], *bo[2], *ln2_g[2], *ln2_b[2], *b1[2], *b2[2];
  const bf16  *Wqkv[2], *Wo[2], *W1[2], *W2[2];
  const float* pool_q;
  const float* cross_Wq;       // fp32 rows [0:256) of cross_Wqkv (matvec path)
  const float* cross_bqkv;
  const bf16*  cross_Wkv;      // bf16 rows [256:768) of cross_Wqkv
  const float* cross_Wo;
  const float* cross_bo;
  const float *pool_ln_g, *pool_ln_b, *out_ln_g, *out_ln_b;
  const float *triple_pos, *role_pos;
  float* out;
};

// ---- WMMA tile GEMM, G=4 sequences sharing each B fragment ----
// A: bf16 LDS, GSEQ stacked 16-row tiles, leading dim LDA.
// W: (Ncols x K) bf16 row-major (so B = W^T per tile, lane n reads W row n).
// MODE 0: outBf[row*1024+col] = acc+bias (bf16)
// MODE 1: outBf[row*1024+col] = relu(acc+bias) (bf16)
// MODE 2: outF [row*256 +col] += acc+bias (f32 residual)
template <int MODE, int K, int LDA>
__device__ __forceinline__ void gemm_tiles(const bf16* __restrict__ A0,
                                           const bf16* __restrict__ W,
                                           int Ncols,
                                           const float* __restrict__ bias, int bias_off,
                                           bf16* __restrict__ outBf,
                                           float* __restrict__ outF,
                                           int wave, int lane) {
  const int nt    = Ncols >> 4;
  const int nlo   = lane & 15;
  const int khalf = lane >> 4;
  for (int t = wave; t < nt; t += NWAVES) {
    const int   ncb  = t << 4;
    const bf16* wrow = W + (size_t)(ncb + nlo) * K + khalf * 16;
    __builtin_prefetch(wrow + NWAVES * 16 * K, 0, 1);   // next tile -> global_prefetch_b8
    v8f acc[GSEQ] = {};
#pragma unroll 4
    for (int k0 = 0; k0 < K; k0 += 32) {
      v16bf bfr = *(const v16bf*)(wrow + k0);           // one B load ...
#pragma unroll
      for (int g = 0; g < GSEQ; ++g) {                  // ... reused by 4 WMMAs
        v16bf afr = *(const v16bf*)(A0 + (g * 16 + nlo) * LDA + khalf * 16 + k0);
        acc[g] = __builtin_amdgcn_wmma_f32_16x16x32_bf16(false, afr, false, bfr,
                                                         (short)0, acc[g], false, false);
      }
    }
    const int   col = ncb + nlo;
    const int   rb  = khalf << 3;     // C layout: VGPR j -> row j (lanes 0-15) / j+8
    const float bv  = bias[bias_off + col];
#pragma unroll
    for (int g = 0; g < GSEQ; ++g) {
#pragma unroll
      for (int j = 0; j < 8; ++j) {
        const int row = g * 16 + rb + j;
        float v = acc[g][j] + bv;
        if (MODE == 1) v = fmaxf(v, 0.f);
        if (MODE == 2) outF[row * DDIM + col] += v;
        else           outBf[row * 1024 + col] = (bf16)v;
      }
    }
  }
}

// LayerNorm valid rows of Xf (GSEQ x 16 x 256 f32) -> bf16 Abf (stride 256).
__device__ __forceinline__ void ln_to_abf(const float* __restrict__ Xf,
                                          const float* __restrict__ g_,
                                          const float* __restrict__ b_,
                                          bf16* __restrict__ Abf,
                                          float* rowm, float* rowr, int tid) {
  if (tid < GSEQ * SDIM) {
    const int gg = tid / SDIM, i = tid % SDIM, row = gg * 16 + i;
    const float* xr = Xf + row * DDIM;
    float s = 0.f, ss = 0.f;
    for (int c = 0; c < DDIM; ++c) { float v = xr[c]; s += v; ss += v * v; }
    float mean = s * (1.f / DDIM);
    float var  = ss * (1.f / DDIM) - mean * mean;
    rowm[row] = mean;
    rowr[row] = rsqrtf(var + EPSV);
  }
  __syncthreads();
  for (int idx = tid; idx < GSEQ * 16 * DDIM; idx += NTHR) {
    int row = idx >> 8, col = idx & 255, rl = row & 15;
    float v = 0.f;
    if (rl < SDIM) v = (Xf[idx] - rowm[row]) * rowr[row] * g_[col] + b_[col];
    Abf[idx] = (bf16)v;
  }
  __syncthreads();
}

extern __shared__ char smem_raw[];

__global__ __launch_bounds__(NTHR) void triple_compressor_seq(SeqParams p) {
  const int tid  = threadIdx.x;
  const int lane = tid & 31;
  const int wave = tid >> 5;
  const int seq0 = blockIdx.x * GSEQ;

  float* Xf     = (float*)smem_raw;             // GSEQ*16 x 256 f32 residual (64KB)
  bf16*  Hbf    = (bf16*)(Xf + GSEQ * 16 * DDIM);   // GSEQ*16 x 1024 bf16 scratch (128KB)
  bf16*  Abf    = Hbf + GSEQ * 16 * 1024;       // GSEQ*16 x 256 bf16 A staging (32KB)
  float* rowm   = (float*)(Abf + GSEQ * 16 * DDIM); // 64
  float* rowr   = rowm + 64;                    // 64
  float* attw   = rowr + 64;                    // GSEQ*4*16
  float* qvec   = attw + GSEQ * 64;             // GSEQ*256
  float* red    = qvec + GSEQ * DDIM;           // 256
  float* red2   = red + NTHR;                   // 256

  // ---- embedding gather + positional add (4 sequences) ----
  for (int idx = tid; idx < GSEQ * 16 * DDIM; idx += NTHR) {
    int row = idx >> 8, col = idx & 255, g = row >> 4, rl = row & 15;
    float v = 0.f;
    if (rl < SDIM) {
      long long id = p.token_ids[(size_t)(seq0 + g) * SDIM + rl];
      v = p.token_emb[id * (long long)DDIM + col] + p.token_pos[rl * DDIM + col];
    }
    Xf[idx] = v;
  }
  __syncthreads();

  // ---- encoder layers ----
  for (int l = 0; l < 2; ++l) {
    ln_to_abf(Xf, p.ln1_g[l], p.ln1_b[l], Abf, rowm, rowr, tid);
    // QKV projection: Hbf[:, 0:768] (bf16, bias fused)
    gemm_tiles<0, DDIM, DDIM>(Abf, p.Wqkv[l], 3 * DDIM, p.bqkv[l], 0, Hbf, nullptr, wave, lane);
    __syncthreads();
    // masked MHA: GSEQ*4 heads * 12 queries = 192 tuples
    for (int pidx = tid; pidx < GSEQ * SDIM * HHEADS; pidx += NTHR) {
      int g = pidx / (SDIM * HHEADS), rem = pidx % (SDIM * HHEADS);
      int i = rem >> 2, h = rem & 3;
      const uint8_t* km   = p.tok_mask + (size_t)(seq0 + g) * SDIM;
      const bf16*    qrow = Hbf + (g * 16 + i) * 1024 + h * DHEAD;
      float sc[SDIM]; float mx = -1e30f;
      for (int j = 0; j < SDIM; ++j) {
        const bf16* krow = Hbf + (g * 16 + j) * 1024 + 256 + h * DHEAD;
        float s = 0.f;
        for (int d = 0; d < DHEAD; ++d) s += (float)qrow[d] * (float)krow[d];
        s *= 0.125f;
        if (km[j]) s = NEGV;
        sc[j] = s; mx = fmaxf(mx, s);
      }
      float sum = 0.f;
      for (int j = 0; j < SDIM; ++j) { sc[j] = __expf(sc[j] - mx); sum += sc[j]; }
      float inv = 1.f / sum;
      for (int d = 0; d < DHEAD; ++d) {
        float o = 0.f;
        for (int j = 0; j < SDIM; ++j)
          o += sc[j] * (float)Hbf[(g * 16 + j) * 1024 + 512 + h * DHEAD + d];
        Abf[(g * 16 + i) * DDIM + h * DHEAD + d] = (bf16)(o * inv);
      }
    }
    __syncthreads();
    // output projection + residual into Xf
    gemm_tiles<2, DDIM, DDIM>(Abf, p.Wo[l], DDIM, p.bo[l], 0, nullptr, Xf, wave, lane);
    __syncthreads();
    // FFN: LN2 -> Abf ; relu(*W1^T+b1) fused -> Hbf bf16 ; *W2^T+b2 residual
    ln_to_abf(Xf, p.ln2_g[l], p.ln2_b[l], Abf, rowm, rowr, tid);
    gemm_tiles<1, DDIM, DDIM>(Abf, p.W1[l], FFDIM, p.b1[l], 0, Hbf, nullptr, wave, lane);
    __syncthreads();
    gemm_tiles<2, FFDIM, 1024>(Hbf, p.W2[l], DDIM, p.b2[l], 0, nullptr, Xf, wave, lane);
    __syncthreads();
  }

  // ---- cross-attention pooling ----
  for (int idx = tid; idx < GSEQ * 16 * DDIM; idx += NTHR) {
    int rl = (idx >> 8) & 15;
    Abf[idx] = (bf16)((rl < SDIM) ? Xf[idx] : 0.f);
  }
  __syncthreads();
  // K|V projection: Hbf[:, 0:512] (bias rows 256..768 of cross_bqkv)
  gemm_tiles<0, DDIM, DDIM>(Abf, p.cross_Wkv, 2 * DDIM, p.cross_bqkv, DDIM, Hbf, nullptr, wave, lane);
  // q_g = pool_q[r_g] @ Wq^T + bq  (fp32 matvec, GSEQ*256 outputs)
  for (int o = tid; o < GSEQ * DDIM; o += NTHR) {
    int g = o >> 8, oc = o & 255;
    int rg = (seq0 + g) % RDIM;
    const float* pq = p.pool_q + rg * DDIM;
    const float* wr = p.cross_Wq + (size_t)oc * DDIM;
    float s = p.cross_bqkv[oc];
    for (int k = 0; k < DDIM; ++k) s += pq[k] * wr[k];
    qvec[o] = s;
  }
  __syncthreads();
  if (tid < GSEQ * HHEADS) {
    int g = tid >> 2, h = tid & 3;
    const uint8_t* km = p.tok_mask + (size_t)(seq0 + g) * SDIM;
    float sc[SDIM]; float mx = -1e30f;
    for (int j = 0; j < SDIM; ++j) {
      float s = 0.f;
      for (int d = 0; d < DHEAD; ++d)
        s += qvec[g * DDIM + h * DHEAD + d] * (float)Hbf[(g * 16 + j) * 1024 + h * DHEAD + d];
      s *= 0.125f;
      if (km[j]) s = NEGV;
      sc[j] = s; mx = fmaxf(mx, s);
    }
    float sum = 0.f;
    for (int j = 0; j < SDIM; ++j) { sc[j] = __expf(sc[j] - mx); sum += sc[j]; }
    float inv = 1.f / sum;
    for (int j = 0; j < SDIM; ++j) attw[(g * HHEADS + h) * 16 + j] = sc[j] * inv;
  }
  __syncthreads();
  // attn-weighted V -> red2 staging? No: write pooled into qvec after use.
  // pooled_g[d] = sum_j attw * V ; then out-proj matvec. Use red/red2 later only.
  {
    // pooled -> stored temporarily in place of attw? keep separate: reuse Xf row 0 area is
    // unsafe; write into qvec only after reading it. Stage pooled in registers via two passes:
    // pass 1 compute pooled into red-free LDS region: reuse Abf as f32? Abf is bf16.
    // Simplest: store pooled into the first GSEQ*256 floats of Xf (residual no longer needed).
  }
  float* pooled = Xf;                    // residual dead from here; reuse as GSEQ*256 f32
  __syncthreads();
  for (int o = tid; o < GSEQ * DDIM; o += NTHR) {
    int g = o >> 8, d = o & 255, h = d >> 6;
    float v = 0.f;
    for (int j = 0; j < SDIM; ++j)
      v += attw[(g * HHEADS + h) * 16 + j] * (float)Hbf[(g * 16 + j) * 1024 + 256 + d];
    pooled[g * DDIM + d] = v;
  }
  __syncthreads();
  // out-proj matvec (reads pooled, writes qvec; qvec scores already consumed)
  for (int o = tid; o < GSEQ * DDIM; o += NTHR) {
    int g = o >> 8, oc = o & 255;
    const float* wr = p.cross_Wo + (size_t)oc * DDIM;
    float s = p.cross_bo[oc];
    for (int k = 0; k < DDIM; ++k) s += pooled[g * DDIM + k] * wr[k];
    qvec[o] = s;
  }
  __syncthreads();
  // pool-LN + mask + out-LN + positional adds; 64-thread group per sequence.
  {
    const int gg = tid >> 6, t64 = tid & 63;
    const int sg = seq0 + gg;
    const int rg = sg % RDIM, mg = (sg / RDIM) % MDIM, bg = sg / (RDIM * MDIM);
    float ls = 0.f, lss = 0.f;
    for (int d = t64; d < DDIM; d += 64) { float v = qvec[gg * DDIM + d]; ls += v; lss += v * v; }
    red[tid] = ls; red2[tid] = lss;
    __syncthreads();
    for (int off = 32; off > 0; off >>= 1) {
      if (t64 < off) { red[tid] += red[tid + off]; red2[tid] += red2[tid + off]; }
      __syncthreads();
    }
    float mean = red[gg * 64] * (1.f / DDIM);
    float var  = red2[gg * 64] * (1.f / DDIM) - mean * mean;
    float rs   = rsqrtf(var + EPSV);
    const bool valid = (p.trip_mask[bg * MDIM + mg] == 0);
    // thread-local: same element->thread map for pool-LN and out-LN
    float vr[4]; float ls2 = 0.f, lss2 = 0.f;
    int c = 0;
    for (int d = t64; d < DDIM; d += 64, ++c) {
      float v = (qvec[gg * DDIM + d] - mean) * rs * p.pool_ln_g[d] + p.pool_ln_b[d];
      v = valid ? v : 0.f;               // zero padded-triple rows before out-LN
      vr[c] = v; ls2 += v; lss2 += v * v;
    }
    __syncthreads();                      // red reads above done before overwrite
    red[tid] = ls2; red2[tid] = lss2;
    __syncthreads();
    for (int off = 32; off > 0; off >>= 1) {
      if (t64 < off) { red[tid] += red[tid + off]; red2[tid] += red2[tid + off]; }
      __syncthreads();
    }
    float mean2 = red[gg * 64] * (1.f / DDIM);
    float var2  = red2[gg * 64] * (1.f / DDIM) - mean2 * mean2;
    float rs2   = rsqrtf(var2 + EPSV);
    float* orow = p.out + (size_t)sg * DDIM;
    c = 0;
    for (int d = t64; d < DDIM; d += 64, ++c) {
      orow[d] = (vr[c] - mean2) * rs2 * p.out_ln_g[d] + p.out_ln_b[d]
              + p.triple_pos[mg * DDIM + d] + p.role_pos[rg * DDIM + d];
    }
  }
}

__global__ void cvt_f32_to_bf16(const float* __restrict__ src, bf16* __restrict__ dst, int n) {
  int i = blockIdx.x * blockDim.x + threadIdx.x;
  if (i < n) dst[i] = (bf16)src[i];
}

extern "C" void kernel_launch(void* const* d_in, const int* in_sizes, int n_in,
                              void* d_out, int out_size, void* d_ws, size_t ws_size,
                              hipStream_t stream) {
  (void)in_sizes; (void)n_in; (void)out_size; (void)ws_size;
  auto fp = [&](int i) { return (const float*)d_in[i]; };
  bf16* wsb = (bf16*)d_ws;

  // ---- one-pass weight conversion to bf16 (deterministic, every call) ----
  auto cvt = [&](const float* src, bf16* dst, int n) {
    cvt_f32_to_bf16<<<(n + 255) / 256, 256, 0, stream>>>(src, dst, n);
  };
  for (int l = 0; l < 2; ++l) {
    const int base = 5 + l * 12;
    bf16* lb = wsb + (size_t)l * WS_LAYER;
    cvt(fp(base + 2),  lb + WS_WQKV, 768 * 256);   // Wqkv
    cvt(fp(base + 4),  lb + WS_WO,   256 * 256);   // Wo
    cvt(fp(base + 8),  lb + WS_W1,  1024 * 256);   // W1
    cvt(fp(base + 10), lb + WS_W2,  256 * 1024);   // W2
  }
  cvt(fp(30) + 256 * 256, wsb + WS_CWKV, 512 * 256);  // cross_Wqkv rows [256:768)

  // ---- main kernel: 4 sequences / workgroup ----
  SeqParams p;
  p.token_ids = (const int*)d_in[0];
  p.tok_mask  = (const uint8_t*)d_in[1];
  p.trip_mask = (const uint8_t*)d_in[2];
  p.token_emb = fp(3);
  p.token_pos = fp(4);
  for (int l = 0; l < 2; ++l) {
    const int base = 5 + l * 12;
    bf16* lb = wsb + (size_t)l * WS_LAYER;
    p.ln1_g[l] = fp(base + 0);  p.ln1_b[l] = fp(base + 1);
    p.Wqkv[l]  = lb + WS_WQKV;  p.bqkv[l]  = fp(base + 3);
    p.Wo[l]    = lb + WS_WO;    p.bo[l]    = fp(base + 5);
    p.ln2_g[l] = fp(base + 6);  p.ln2_b[l] = fp(base + 7);
    p.W1[l]    = lb + WS_W1;    p.b1[l]    = fp(base + 9);
    p.W2[l]    = lb + WS_W2;    p.b2[l]    = fp(base + 11);
  }
  p.pool_q     = fp(29);
  p.cross_Wq   = fp(30);
  p.cross_bqkv = fp(31);
  p.cross_Wkv  = wsb + WS_CWKV;
  p.cross_Wo   = fp(32);
  p.cross_bo   = fp(33);
  p.pool_ln_g  = fp(34); p.pool_ln_b = fp(35);
  p.out_ln_g   = fp(36); p.out_ln_b  = fp(37);
  p.triple_pos = fp(38); p.role_pos  = fp(39);
  p.out        = (float*)d_out;

  const size_t smem = (size_t)GSEQ * 16 * DDIM * sizeof(float)   // Xf
                    + (size_t)GSEQ * 16 * 1024 * sizeof(bf16)    // Hbf
                    + (size_t)GSEQ * 16 * DDIM * sizeof(bf16)    // Abf
                    + (64 + 64 + GSEQ * 64 + GSEQ * DDIM + NTHR + NTHR) * sizeof(float);
  triple_compressor_seq<<<NSEQ / GSEQ, NTHR, smem, stream>>>(p);
}
